// BlipAttention_27401891349298
// MI455X (gfx1250) — matrix operand order repaired and checked by reading, not compiled, not run on hardware
//
#include <hip/hip_runtime.h>
#include <hip/hip_bf16.h>
#include <stdint.h>

#define EMBED   1408
#define NH      16
#define HD      88
#define HDP     96
#define BATCH   16
#define SEQ     577
#define TOK     (BATCH*SEQ)      /* 9232 */
#define SPAD    640              /* 10*64, padded key/query rows */
#define NKCH    10               /* key chunks of 64 */
#define SCALE_F 0.1066003581778052f  /* 88^-0.5 */

typedef __attribute__((ext_vector_type(16))) __bf16       v16bf;
typedef __attribute__((ext_vector_type(8)))  float        v8f;
typedef __attribute__((ext_vector_type(8)))  unsigned int v8u;

// CDNA5 split workgroup barrier: signal after LDS stores drain, do useful
// work (issue next tile's global loads), then wait. Fallback: __syncthreads.
#if defined(__has_builtin)
#if __has_builtin(__builtin_amdgcn_s_barrier_signal) && \
    __has_builtin(__builtin_amdgcn_s_barrier_wait) &&   \
    __has_builtin(__builtin_amdgcn_s_wait_dscnt)
#define USE_SPLIT_BARRIER 1
#endif
#endif
#ifndef USE_SPLIT_BARRIER
#define USE_SPLIT_BARRIER 0
#endif

static __device__ __forceinline__ void tile_signal() {
#if USE_SPLIT_BARRIER
  __builtin_amdgcn_s_wait_dscnt(0);      // my LDS stores are visible
  __builtin_amdgcn_s_barrier_signal(-1); // arrive at workgroup barrier
#endif
}
static __device__ __forceinline__ void tile_wait() {
#if USE_SPLIT_BARRIER
  __builtin_amdgcn_s_barrier_wait(-1);   // everyone's stores visible
#else
  __syncthreads();
#endif
}

static __device__ __forceinline__ unsigned short f2bf(float f) {
  unsigned int u = __builtin_bit_cast(unsigned int, f);
  u += 0x7fffu + ((u >> 16) & 1u);   // round-to-nearest-even
  return (unsigned short)(u >> 16);
}

static __device__ __forceinline__ v8f wmma_bf16(v8u a, v8u b, v8f c) {
  return __builtin_amdgcn_wmma_f32_16x16x32_bf16(
      false, __builtin_bit_cast(v16bf, a),
      false, __builtin_bit_cast(v16bf, b),
      (short)0, c, false, false);
}

static __device__ __forceinline__ v8f zero8() {
  v8f z = {0.f,0.f,0.f,0.f,0.f,0.f,0.f,0.f};
  return z;
}

// ---------------------------------------------------------------- convert
// n4 = element-count/4 (all buffers here are multiples of 4)
__global__ __launch_bounds__(256)
void f32_to_bf16_kernel(const float* __restrict__ in,
                        unsigned short* __restrict__ out, int n4) {
  int i = blockIdx.x * 256 + threadIdx.x;
  if (i < n4) {
    const float4 v = *(const float4*)(in + (size_t)i * 4);
    uint2 p;
    p.x = (unsigned int)f2bf(v.x) | ((unsigned int)f2bf(v.y) << 16);
    p.y = (unsigned int)f2bf(v.z) | ((unsigned int)f2bf(v.w) << 16);
    *(uint2*)(out + (size_t)i * 4) = p;
  }
}

// ---------------------------------------------------------------- tiled GEMM
// C[M,N] = A[M,K](bf16) * B[K,N](bf16) + bias, software-pipelined tiles.
// EPI==0: QKV epilogue (split q|k|v, scale q, scatter to [b,h,SPAD,88] bf16)
// EPI==1: fp32 + bias to f_out[M,N]
template<int EPI>
__global__ __launch_bounds__(256)
void gemm_bf16_kernel(const unsigned short* __restrict__ A,
                      const unsigned short* __restrict__ Bw,
                      const float* __restrict__ bias,
                      int M, int N, int K,
                      unsigned short* __restrict__ q_out,
                      unsigned short* __restrict__ k_out,
                      unsigned short* __restrict__ v_out,
                      float* __restrict__ f_out) {
  __shared__ unsigned short Al[128][40];   // [m][k], 80B row stride
  __shared__ unsigned short Btl[128][40];  // [n][k] (transposed B tile)

  const int tid  = threadIdx.x;
  const int lane = tid & 31;
  const int wid  = tid >> 5;
  const int wm   = wid >> 1;        // 0..3 : 32-row slice
  const int wn   = wid & 1;         // 0..1 : 64-col slice
  const int m0   = blockIdx.x * 128;
  const int n0   = blockIdx.y * 128;
  const int ln   = lane & 15;
  const int lh   = lane >> 4;

  v8f acc[2][4];
  for (int i = 0; i < 2; ++i)
    for (int j = 0; j < 4; ++j) acc[i][j] = zero8();

  const int nk = K >> 5;

  uint4 areg[2];    // A-tile stage: 2x uint4 per thread
  uint4 breg[8];    // B-tile stage: 8x uint4 (threads 0..63 only)

  auto load_tiles = [&](int kk) {
    const int k0 = kk << 5;
    #pragma unroll
    for (int t = 0; t < 2; ++t) {
      const int i   = tid + t * 256;
      const int row = i >> 2;
      const int col = (i & 3) << 3;
      uint4 v = make_uint4(0u, 0u, 0u, 0u);
      if (m0 + row < M) {
        const unsigned short* p = A + (size_t)(m0 + row) * K + k0 + col;
        v = *(const uint4*)p;
        if (kk + 1 < nk) __builtin_prefetch(p + 32, 0, 0);  // tile after next
      }
      areg[t] = v;
    }
    if (tid < 64) {
      const int kr  = (tid >> 4) << 3;
      const int col = (tid & 15) << 3;
      #pragma unroll
      for (int r = 0; r < 8; ++r)
        breg[r] = *(const uint4*)(Bw + (size_t)(k0 + kr + r) * N + n0 + col);
    }
  };

  auto store_tiles = [&]() {
    #pragma unroll
    for (int t = 0; t < 2; ++t) {
      const int i = tid + t * 256;
      *(uint4*)&Al[i >> 2][(i & 3) << 3] = areg[t];
    }
    if (tid < 64) {
      const int kr  = (tid >> 4) << 3;
      const int col = (tid & 15) << 3;
      unsigned short buf[8][8];
      #pragma unroll
      for (int r = 0; r < 8; ++r) *(uint4*)&buf[r][0] = breg[r];
      #pragma unroll
      for (int j = 0; j < 8; ++j) {     // 8x8 register transpose
        uint4 w;
        w.x = (unsigned int)buf[0][j] | ((unsigned int)buf[1][j] << 16);
        w.y = (unsigned int)buf[2][j] | ((unsigned int)buf[3][j] << 16);
        w.z = (unsigned int)buf[4][j] | ((unsigned int)buf[5][j] << 16);
        w.w = (unsigned int)buf[6][j] | ((unsigned int)buf[7][j] << 16);
        *(uint4*)&Btl[col + j][kr] = w;
      }
    }
  };

  load_tiles(0);
  for (int kk = 0; kk < nk; ++kk) {
    __syncthreads();                 // prior compute done reading LDS
    store_tiles();
    tile_signal();                   // s_wait_dscnt 0 + s_barrier_signal -1
    if (kk + 1 < nk) load_tiles(kk + 1);   // HBM loads fly during barrier
    tile_wait();                     // s_barrier_wait -1

    // A fragments (k order used here, matching B: lane half h holds k 16h..16h+15)
    v8u af[2];
    #pragma unroll
    for (int mt = 0; mt < 2; ++mt) {
      const unsigned short* row = Al[wm * 32 + mt * 16 + ln];
      const int kb = lh << 4;
      v8u a;
      #pragma unroll
      for (int j = 0; j < 4; ++j) a[j]     = *(const unsigned int*)(row + kb + 2 * j);
      #pragma unroll
      for (int j = 0; j < 4; ++j) a[4 + j] = *(const unsigned int*)(row + kb + 8 + 2 * j);
      af[mt] = a;
    }
    // B fragments (k order used here, matching A: lane half h holds k 16h..16h+15)
    #pragma unroll
    for (int nt = 0; nt < 4; ++nt) {
      const unsigned short* row = Btl[wn * 64 + nt * 16 + ln];
      const int kb = lh << 4;
      v8u b;
      #pragma unroll
      for (int j = 0; j < 8; ++j) b[j] = *(const unsigned int*)(row + kb + 2 * j);
      #pragma unroll
      for (int mt = 0; mt < 2; ++mt)
        acc[mt][nt] = wmma_bf16(af[mt], b, acc[mt][nt]);
    }
  }

  // epilogue: C/D layout row = base + r + 8*lane_half, col = base + (lane&15)
  #pragma unroll
  for (int nt = 0; nt < 4; ++nt) {
    const int gn = n0 + wn * 64 + nt * 16 + ln;
    const float bs = bias[gn];
    #pragma unroll
    for (int mt = 0; mt < 2; ++mt) {
      #pragma unroll
      for (int r = 0; r < 8; ++r) {
        const int gm = m0 + wm * 32 + mt * 16 + r + (lh << 3);
        if (gm >= M) continue;
        float val = acc[mt][nt][r] + bs;
        if (EPI == 0) {
          const int sel = gn / EMBED;
          const int d   = gn - sel * EMBED;
          const int h   = d / HD;
          const int hd  = d - h * HD;
          if (sel == 0) val *= SCALE_F;
          const int b = gm / SEQ;
          const int s = gm - b * SEQ;
          unsigned short* dst = (sel == 0) ? q_out : ((sel == 1) ? k_out : v_out);
          dst[((size_t)(b * NH + h) * SPAD + s) * HD + hd] = f2bf(val);
        } else {
          f_out[(size_t)gm * N + gn] = val;
        }
      }
    }
  }
}

// ---------------------------------------------------------------- flash attention
// Q,K,V: bf16 [B*H][SPAD][88] (Q pre-scaled, pad rows zero). O: bf16 [B*S][1408].
// Ones column planted at head-dim 88 of the V tile accumulates the softmax
// denominator inside the same WMMA chain (augmented-matrix trick).
__global__ __launch_bounds__(256)
void attn_kernel(const unsigned short* __restrict__ Q,
                 const unsigned short* __restrict__ Kmat,
                 const unsigned short* __restrict__ Vmat,
                 unsigned short* __restrict__ O) {
  __shared__ unsigned short Kl[64][104];   // [key][d], 208B stride
  __shared__ unsigned short Vt[HDP][72];   // [d][key], 144B stride
  __shared__ unsigned short Pl[8][16][72]; // per-wave probs staging

  const int tid  = threadIdx.x;
  const int lane = tid & 31;
  const int wid  = tid >> 5;
  const int bh   = blockIdx.x / 5;
  const int qblk = blockIdx.x - bh * 5;
  int qt = qblk * 8 + wid;                 // 16-query tile index
  if (qt > 36) qt = 36;                    // duplicated tile, stores identical
  const int ln = lane & 15;
  const int lh = lane >> 4;

  // Q fragments: 3 K-steps over padded head dim (88 -> 96)
  v8u qf[3];
  {
    const int row = qt * 16 + ln;          // < 640, pad rows are zeroed
    const unsigned short* qrow = Q + ((size_t)bh * SPAD + row) * HD;
    const int kb = lh << 4;
    #pragma unroll
    for (int ks = 0; ks < 3; ++ks) {
      v8u a;
      #pragma unroll
      for (int j = 0; j < 4; ++j) {
        const int d = ks * 32 + kb + 2 * j;
        a[j] = (d < HD) ? *(const unsigned int*)(qrow + d) : 0u;
      }
      #pragma unroll
      for (int j = 0; j < 4; ++j) {
        const int d = ks * 32 + kb + 8 + 2 * j;
        a[4 + j] = (d < HD) ? *(const unsigned int*)(qrow + d) : 0u;
      }
      qf[ks] = a;
    }
  }

  float mrow[8];
  v8f o[6];
  #pragma unroll
  for (int r = 0; r < 8; ++r) mrow[r] = -1e30f;
  #pragma unroll
  for (int t = 0; t < 6; ++t) o[t] = zero8();

  uint2 kreg[6], vreg[6];   // pipelined K/V chunk stage (6x uint2 per thread)

  auto load_chunk = [&](int c) {
    #pragma unroll
    for (int t = 0; t < 6; ++t) {
      const int i   = tid + t * 256;
      const int key = i / 24;
      const int d   = (i - key * 24) * 4;
      uint2 kv = make_uint2(0u, 0u), vv = make_uint2(0u, 0u);
      if (d < HD) {
        const size_t base = ((size_t)bh * SPAD + c * 64 + key) * HD + d;
        kv = *(const uint2*)(Kmat + base);
        vv = *(const uint2*)(Vmat + base);
      }
      kreg[t] = kv; vreg[t] = vv;
    }
  };
  auto store_chunk = [&]() {
    #pragma unroll
    for (int t = 0; t < 6; ++t) {
      const int i   = tid + t * 256;
      const int key = i / 24;
      const int d   = (i - key * 24) * 4;
      uint2 kv = kreg[t], vv = vreg[t];
      if (d == HD) vv.x = 0x3F80u;           // bf16 1.0 denominator column
      *(uint2*)&Kl[key][d] = kv;
      Vt[d    ][key] = (unsigned short)(vv.x & 0xffffu);
      Vt[d + 1][key] = (unsigned short)(vv.x >> 16);
      Vt[d + 2][key] = (unsigned short)(vv.y & 0xffffu);
      Vt[d + 3][key] = (unsigned short)(vv.y >> 16);
    }
  };

  load_chunk(0);
  for (int c = 0; c < NKCH; ++c) {
    __syncthreads();                 // prior chunk's LDS reads done
    store_chunk();
    tile_signal();
    if (c + 1 < NKCH) load_chunk(c + 1);  // next chunk in flight over barrier
    tile_wait();

    // scores: S_tile = Q(16x96) * K^T(96x64) as four 16-key subtiles
    v8f s[4];
    #pragma unroll
    for (int st = 0; st < 4; ++st) {
      v8f a = zero8();
      const int key = st * 16 + ln;
      const int kb  = lh << 4;
      #pragma unroll
      for (int ks = 0; ks < 3; ++ks) {
        v8u b;
        #pragma unroll
        for (int j = 0; j < 8; ++j)
          b[j] = *(const unsigned int*)&Kl[key][ks * 32 + kb + 2 * j];
        a = wmma_bf16(qf[ks], b, a);
      }
      const int gkey = c * 64 + st * 16 + ln;
      if (gkey >= SEQ) {
        #pragma unroll
        for (int r = 0; r < 8; ++r) a[r] = -1e30f;
      }
      s[st] = a;
    }

    // online softmax max: lane-local reduce then batched xor-shuffle steps
    float mx[8];
    #pragma unroll
    for (int r = 0; r < 8; ++r)
      mx[r] = fmaxf(fmaxf(s[0][r], s[1][r]), fmaxf(s[2][r], s[3][r]));
    #pragma unroll
    for (int off = 8; off > 0; off >>= 1) {
      #pragma unroll
      for (int r = 0; r < 8; ++r)
        mx[r] = fmaxf(mx[r], __shfl_xor(mx[r], off, 32));
    }
    float scl[8];
    #pragma unroll
    for (int r = 0; r < 8; ++r) {
      const float nm = fmaxf(mrow[r], mx[r]);
      scl[r] = __expf(mrow[r] - nm);
      mrow[r] = nm;
      #pragma unroll
      for (int st = 0; st < 4; ++st) s[st][r] = __expf(s[st][r] - nm);
    }
    #pragma unroll
    for (int t = 0; t < 6; ++t)
      #pragma unroll
      for (int r = 0; r < 8; ++r) o[t][r] *= scl[r];

    // C-layout -> A-layout via per-wave LDS staging
    #pragma unroll
    for (int st = 0; st < 4; ++st)
      #pragma unroll
      for (int r = 0; r < 8; ++r)
        Pl[wid][r + (lh << 3)][st * 16 + ln] = f2bf(s[st][r]);

    v8u pf[2];
    {
      const unsigned short* prow = Pl[wid][ln];
      const int kb = lh << 4;
      #pragma unroll
      for (int ks = 0; ks < 2; ++ks) {
        v8u p;
        #pragma unroll
        for (int j = 0; j < 4; ++j) p[j]     = *(const unsigned int*)(prow + ks * 32 + kb + 2 * j);
        #pragma unroll
        for (int j = 0; j < 4; ++j) p[4 + j] = *(const unsigned int*)(prow + ks * 32 + kb + 8 + 2 * j);
        pf[ks] = p;
      }
    }
    // O += P(16x64) * V(64x96)  (V col 88 == ones -> running denominator)
    #pragma unroll
    for (int t = 0; t < 6; ++t) {
      const unsigned short* vrow = Vt[t * 16 + ln];
      const int kb = lh << 4;
      #pragma unroll
      for (int ks = 0; ks < 2; ++ks) {
        v8u b;
        #pragma unroll
        for (int j = 0; j < 8; ++j)
          b[j] = *(const unsigned int*)(vrow + ks * 32 + kb + 2 * j);
        o[t] = wmma_bf16(pf[ks], b, o[t]);
      }
    }
  }

  // denominator lives in o[5] column 88 (ln==8): broadcast within lane half
  const int srcl = (lane & 16) + 8;
  const int b = bh >> 4;
  const int h = bh & 15;
  #pragma unroll
  for (int r = 0; r < 8; ++r) {
    const float lsum = __shfl(o[5][r], srcl, 32);
    const float inv  = 1.f / lsum;
    const int row = qt * 16 + r + (lh << 3);
    if (row >= SEQ) continue;
    #pragma unroll
    for (int t = 0; t < 6; ++t) {
      const int d = t * 16 + ln;
      if (d < HD)
        O[((size_t)(b * SEQ + row)) * EMBED + h * HD + d] = f2bf(o[t][r] * inv);
    }
  }
}

// ---------------------------------------------------------------- launcher
extern "C" void kernel_launch(void* const* d_in, const int* in_sizes, int n_in,
                              void* d_out, int out_size, void* d_ws, size_t ws_size,
                              hipStream_t stream) {
  (void)in_sizes; (void)n_in; (void)out_size; (void)ws_size;
  const float* X     = (const float*)d_in[0];
  const float* Wqkv  = (const float*)d_in[1];
  const float* Bqkv  = (const float*)d_in[2];
  const float* Wproj = (const float*)d_in[3];
  const float* Bproj = (const float*)d_in[4];
  float* out = (float*)d_out;

  char* ws = (char*)d_ws;
  size_t off = 0;
  auto carve = [&](size_t bytes) -> void* {
    void* p = ws + off;
    off = (off + bytes + 255) & ~(size_t)255;
    return p;
  };
  unsigned short* Xbf = (unsigned short*)carve((size_t)TOK * EMBED * 2);
  unsigned short* Wqb = (unsigned short*)carve((size_t)EMBED * 3 * EMBED * 2);
  unsigned short* Wpb = (unsigned short*)carve((size_t)EMBED * EMBED * 2);
  const size_t qkv_elems = (size_t)BATCH * NH * SPAD * HD;
  unsigned short* Qb = (unsigned short*)carve(qkv_elems * 2);
  unsigned short* Kb = (unsigned short*)carve(qkv_elems * 2);
  unsigned short* Vb = (unsigned short*)carve(qkv_elems * 2);
  unsigned short* Ab = (unsigned short*)carve((size_t)TOK * EMBED * 2);

  const int nX4  = TOK * EMBED / 4;
  const int nWq4 = EMBED * 3 * EMBED / 4;
  const int nWp4 = EMBED * EMBED / 4;
  f32_to_bf16_kernel<<<(nX4  + 255) / 256, 256, 0, stream>>>(X,     Xbf, nX4);
  f32_to_bf16_kernel<<<(nWq4 + 255) / 256, 256, 0, stream>>>(Wqkv,  Wqb, nWq4);
  f32_to_bf16_kernel<<<(nWp4 + 255) / 256, 256, 0, stream>>>(Wproj, Wpb, nWp4);

  hipMemsetAsync(Qb, 0, qkv_elems * 2, stream);
  hipMemsetAsync(Kb, 0, qkv_elems * 2, stream);
  hipMemsetAsync(Vb, 0, qkv_elems * 2, stream);

  // QKV projection: M=9232, N=4224, K=1408
  gemm_bf16_kernel<0><<<dim3(73, 33), 256, 0, stream>>>(
      Xbf, Wqb, Bqkv, TOK, 3 * EMBED, EMBED, Qb, Kb, Vb, nullptr);

  // flash attention: 256 (b,h) pairs * 5 query blocks
  attn_kernel<<<BATCH * NH * 5, 256, 0, stream>>>(Qb, Kb, Vb, Ab);

  // output projection: M=9232, N=1408, K=1408
  gemm_bf16_kernel<1><<<dim3(73, 11), 256, 0, stream>>>(
      Ab, Wpb, Bproj, TOK, EMBED, EMBED, nullptr, nullptr, nullptr, out);
}
